// MoELayer_84447646974397
// MI455X (gfx1250) — matrix operand order repaired and checked by reading, not compiled, not run on hardware
//
#include <hip/hip_runtime.h>
#include <math.h>

typedef __attribute__((ext_vector_type(16))) __bf16       v16bf;
typedef __attribute__((ext_vector_type(8)))  float        v8f;
typedef __attribute__((ext_vector_type(8)))  unsigned int v8u;

#define TOKENS   32768
#define H_DIM    1024
#define F_DIM    2048
#define F4_DIM   512
#define NEXP     8
#define CAP      4096
#define NASSIGN  (TOKENS * 2)
#define NBLK     (NASSIGN / 256)
#define EPS_RMS  1e-6f

// ---------------- helpers ----------------

__device__ __forceinline__ __bf16 f2bf(float f) {
  unsigned u = __builtin_bit_cast(unsigned, f);
  unsigned r = (u + 0x7FFFu + ((u >> 16) & 1u)) >> 16;
  unsigned short s = (unsigned short)r;
  return __builtin_bit_cast(__bf16, s);
}

__device__ __forceinline__ unsigned pack_bf16x2(float lo, float hi) {
  unsigned a = (unsigned)__builtin_bit_cast(unsigned short, f2bf(lo));
  unsigned b = (unsigned)__builtin_bit_cast(unsigned short, f2bf(hi));
  return a | (b << 16);
}

__device__ __forceinline__ float gelu_exact(float x) {
  return 0.5f * x * (1.0f + erff(x * 0.70710678118654752f));
}

// A fragment: 16x32 bf16, row-major A, lda = K stride (elements).
// lanes 0-15: M=lane, K in {0..7,16..23}; lanes 16-31: M=lane-16, K in {8..15,24..31}
// Per lane this is two contiguous 16B regions -> coalesced b128 loads.
__device__ __forceinline__ v16bf frag_a(const __bf16* __restrict__ A, long lda,
                                        int m0, int k0, int lane) {
  int r  = lane & 15;
  int hk = (lane >> 4) << 3;            // 0 or 8
  const __bf16* row = A + (long)(m0 + r) * lda + k0;
  v8u u;
#pragma unroll
  for (int i = 0; i < 8; ++i) {
    int k = ((i < 4) ? (2 * i) : (16 + 2 * (i - 4))) + hk;
    u[i] = *(const unsigned*)(row + k);  // bf16 pair (k, k+1)
  }
  return __builtin_bit_cast(v16bf, u);
}

// B fragment from pre-swizzled storage: tile (kt,nt) holds 32 lanes x 8 dwords
// contiguously. One 32-byte load per lane.
__device__ __forceinline__ v16bf frag_b_packed(const __bf16* __restrict__ Bp,
                                               int ntn, int kt, int nt, int lane) {
  const v8u* p = (const v8u*)Bp + ((long)(kt * ntn + nt) * 32 + lane);
  return __builtin_bit_cast(v16bf, *p);
}

__device__ __forceinline__ v8f wmma_bf16(v16bf a, v16bf b, v8f c) {
  return __builtin_amdgcn_wmma_f32_16x16x32_bf16(false, a, false, b,
                                                 (short)0, c, false, false);
}

// ---------------- weight pre-swizzle (f32 [K,N] -> packed bf16 B-fragments) ----
// Packed layout: dword index = ((kt*(N/16)+nt)*32 + lane)*8 + i
//   lane: c=lane&15 -> column nt*16+c ; hk=(lane>>4)*8
//   dword i holds rows k = kt*32 + hk + (i<4 ? 2i : 16+2(i-4)) and k+1 at that column.

__global__ void __launch_bounds__(256)
pack_b_kernel(const float* __restrict__ src, __bf16* __restrict__ dst,
              int N, int K, long strideSrc, long strideDst) {
  long e = blockIdx.z;
  src += e * strideSrc;
  unsigned* dw = (unsigned*)(dst + e * strideDst);
  long idx = (long)blockIdx.x * 256 + threadIdx.x;       // dword index
  long total = (long)K * N / 2;
  if (idx >= total) return;
  int  i    = (int)(idx & 7);
  int  lane = (int)((idx >> 3) & 31);
  long tile = idx >> 8;
  int  ntn  = N >> 4;
  int  nt   = (int)(tile % ntn);
  int  kt   = (int)(tile / ntn);
  int  c    = lane & 15;
  int  hk   = (lane >> 4) << 3;
  int  k    = (kt << 5) + hk + ((i < 4) ? (2 * i) : (16 + 2 * (i - 4)));
  int  n    = (nt << 4) + c;
  float lo = src[(long)k * N + n];
  float hi = src[(long)(k + 1) * N + n];
  dw[idx] = pack_bf16x2(lo, hi);
}

// ---------------- elementwise / routing kernels ----------------

__global__ void __launch_bounds__(256)
router_kernel(const float* __restrict__ x, const float* __restrict__ Wr,
              int* __restrict__ e_flat, float* __restrict__ w_flat) {
  int t = blockIdx.x * blockDim.x + threadIdx.x;
  if (t >= TOKENS) return;
  float acc[NEXP];
#pragma unroll
  for (int e = 0; e < NEXP; ++e) acc[e] = 0.0f;
  const float* xr = x + (long)t * H_DIM;
  for (int h = 0; h < H_DIM; ++h) {
    float xv = xr[h];
    const float* wr = Wr + (long)h * NEXP;
#pragma unroll
    for (int e = 0; e < NEXP; ++e) acc[e] = fmaf(xv, wr[e], acc[e]);
  }
  float m = acc[0];
#pragma unroll
  for (int e = 1; e < NEXP; ++e) m = fmaxf(m, acc[e]);
  float s = 0.0f;
#pragma unroll
  for (int e = 0; e < NEXP; ++e) { acc[e] = __expf(acc[e] - m); s += acc[e]; }
  float inv = 1.0f / s;
#pragma unroll
  for (int e = 0; e < NEXP; ++e) acc[e] *= inv;
  int e0 = 0; float p0 = acc[0];
#pragma unroll
  for (int e = 1; e < NEXP; ++e) if (acc[e] > p0) { p0 = acc[e]; e0 = e; }
  int e1 = -1; float p1 = -1.0f;
#pragma unroll
  for (int e = 0; e < NEXP; ++e)
    if (e != e0 && acc[e] > p1) { p1 = acc[e]; e1 = e; }
  e_flat[2 * t] = e0;     w_flat[2 * t] = p0;
  e_flat[2 * t + 1] = e1; w_flat[2 * t + 1] = p1;
}

__global__ void __launch_bounds__(256)
hist_kernel(const int* __restrict__ e_flat, int* __restrict__ blkhist) {
  __shared__ int cnt[NEXP];
  if (threadIdx.x < NEXP) cnt[threadIdx.x] = 0;
  __syncthreads();
  int i = blockIdx.x * 256 + threadIdx.x;
  atomicAdd(&cnt[e_flat[i]], 1);
  __syncthreads();
  if (threadIdx.x < NEXP) blkhist[blockIdx.x * NEXP + threadIdx.x] = cnt[threadIdx.x];
}

__global__ void __launch_bounds__(32)
scan_kernel(const int* __restrict__ blkhist, int* __restrict__ blkbase) {
  int e = threadIdx.x;
  if (e >= NEXP) return;
  int running = 0;
  for (int b = 0; b < NBLK; ++b) {
    blkbase[b * NEXP + e] = running;
    running += blkhist[b * NEXP + e];
  }
}

__global__ void __launch_bounds__(256)
assign_kernel(const int* __restrict__ e_flat, const int* __restrict__ blkbase,
              int* __restrict__ pos) {
  __shared__ int se[256];
  int i = blockIdx.x * 256 + threadIdx.x;
  int e = e_flat[i];
  se[threadIdx.x] = e;
  __syncthreads();
  int rank = 0;
  for (int j = 0; j < (int)threadIdx.x; ++j) rank += (se[j] == e) ? 1 : 0;
  int p = blkbase[blockIdx.x * NEXP + e] + rank;
  pos[i] = (p < CAP) ? p : -1;
}

__global__ void __launch_bounds__(256)
overflow_kernel(const int* __restrict__ pos, int* __restrict__ ovf) {
  int t = blockIdx.x * blockDim.x + threadIdx.x;
  if (t >= TOKENS) return;
  ovf[t] = (pos[2 * t] < 0 || pos[2 * t + 1] < 0) ? 1 : 0;
}

__global__ void __launch_bounds__(256)
rms_scatter_kernel(const float* __restrict__ x, const float* __restrict__ norm_w,
                   const int* __restrict__ e_flat, const int* __restrict__ pos,
                   __bf16* __restrict__ hbuf) {
  int i = blockIdx.x;               // assignment index
  int p = pos[i];
  if (p < 0) return;
  int e = e_flat[i];
  int t = i >> 1;
  const float* xr = x + (long)t * H_DIM;
  __shared__ float red[8];
  float s = 0.0f;
  for (int h = threadIdx.x; h < H_DIM; h += 256) { float v = xr[h]; s += v * v; }
#pragma unroll
  for (int off = 16; off > 0; off >>= 1) s += __shfl_down(s, off, 32);
  if ((threadIdx.x & 31) == 0) red[threadIdx.x >> 5] = s;
  __syncthreads();
  if (threadIdx.x == 0) {
    float tsum = 0.0f;
#pragma unroll
    for (int w = 0; w < 8; ++w) tsum += red[w];
    red[0] = tsum;
  }
  __syncthreads();
  float scale = rsqrtf(red[0] * (1.0f / H_DIM) + EPS_RMS);
  __bf16* dst = hbuf + ((long)e * CAP + p) * H_DIM;
  const float* nw = norm_w + (long)e * H_DIM;
  for (int h = threadIdx.x; h < H_DIM; h += 256)
    dst[h] = f2bf(xr[h] * scale * nw[h]);
}

__global__ void __launch_bounds__(256)
rms_fallback_kernel(const float* __restrict__ x, const float* __restrict__ f_norm,
                    __bf16* __restrict__ fh) {
  int t = blockIdx.x;
  const float* xr = x + (long)t * H_DIM;
  __shared__ float red[8];
  float s = 0.0f;
  for (int h = threadIdx.x; h < H_DIM; h += 256) { float v = xr[h]; s += v * v; }
#pragma unroll
  for (int off = 16; off > 0; off >>= 1) s += __shfl_down(s, off, 32);
  if ((threadIdx.x & 31) == 0) red[threadIdx.x >> 5] = s;
  __syncthreads();
  if (threadIdx.x == 0) {
    float tsum = 0.0f;
#pragma unroll
    for (int w = 0; w < 8; ++w) tsum += red[w];
    red[0] = tsum;
  }
  __syncthreads();
  float scale = rsqrtf(red[0] * (1.0f / H_DIM) + EPS_RMS);
  __bf16* dst = fh + (long)t * H_DIM;
  for (int h = threadIdx.x; h < H_DIM; h += 256)
    dst[h] = f2bf(xr[h] * scale * f_norm[h]);
}

// ---------------- WMMA GEMM kernels ----------------
// Block tile 64(M) x 128(N); 8 waves, each owns a 32x32 sub-tile.
// A: row-major bf16 [M,K]; Bp: pre-swizzled B fragments; D: row-major.

__global__ void __launch_bounds__(256)
gemm_glu_kernel(const __bf16* __restrict__ A, const __bf16* __restrict__ Bgp,
                const __bf16* __restrict__ Bup, __bf16* __restrict__ D,
                int N, int K, long strideA, long strideB, long strideD) {
  long e = blockIdx.z;
  A  += e * strideA;  Bgp += e * strideB;  Bup += e * strideB;  D += e * strideD;
  int lane = threadIdx.x & 31;
  int wid  = threadIdx.x >> 5;
  int m0 = blockIdx.y * 64  + (wid >> 2) * 32;
  int n0 = blockIdx.x * 128 + (wid & 3) * 32;
  int ntn = N >> 4;
  int nt0 = n0 >> 4;

  v8f accg[2][2] = {{ {0.f}, {0.f} }, { {0.f}, {0.f} }};
  v8f accu[2][2] = {{ {0.f}, {0.f} }, { {0.f}, {0.f} }};

  for (int k0 = 0; k0 < K; k0 += 32) {
    int kt = k0 >> 5;
    v16bf a0  = frag_a(A, K, m0,      k0, lane);
    v16bf a1  = frag_a(A, K, m0 + 16, k0, lane);
    if (k0 + 32 < K)
      __builtin_prefetch(A + (long)(m0 + (lane & 15)) * K + k0 + 32, 0, 0);
    v16bf bg0 = frag_b_packed(Bgp, ntn, kt, nt0,     lane);
    v16bf bg1 = frag_b_packed(Bgp, ntn, kt, nt0 + 1, lane);
    v16bf bu0 = frag_b_packed(Bup, ntn, kt, nt0,     lane);
    v16bf bu1 = frag_b_packed(Bup, ntn, kt, nt0 + 1, lane);
    accg[0][0] = wmma_bf16(a0, bg0, accg[0][0]);
    accg[0][1] = wmma_bf16(a0, bg1, accg[0][1]);
    accg[1][0] = wmma_bf16(a1, bg0, accg[1][0]);
    accg[1][1] = wmma_bf16(a1, bg1, accg[1][1]);
    accu[0][0] = wmma_bf16(a0, bu0, accu[0][0]);
    accu[0][1] = wmma_bf16(a0, bu1, accu[0][1]);
    accu[1][0] = wmma_bf16(a1, bu0, accu[1][0]);
    accu[1][1] = wmma_bf16(a1, bu1, accu[1][1]);
  }

  int c = lane & 15;
  int msel = (lane >> 4) * 8;
#pragma unroll
  for (int mi = 0; mi < 2; ++mi)
#pragma unroll
    for (int ni = 0; ni < 2; ++ni)
#pragma unroll
      for (int r = 0; r < 8; ++r) {
        float g = accg[mi][ni][r];
        float u = accu[mi][ni][r];
        int row = m0 + mi * 16 + msel + r;
        int col = n0 + ni * 16 + c;
        D[(long)row * N + col] = f2bf(gelu_exact(g) * u);
      }
}

__global__ void __launch_bounds__(256)
gemm_f32out_kernel(const __bf16* __restrict__ A, const __bf16* __restrict__ Bp,
                   float* __restrict__ D,
                   int N, int K, long strideA, long strideB, long strideD) {
  long e = blockIdx.z;
  A += e * strideA;  Bp += e * strideB;  D += e * strideD;
  int lane = threadIdx.x & 31;
  int wid  = threadIdx.x >> 5;
  int m0 = blockIdx.y * 64  + (wid >> 2) * 32;
  int n0 = blockIdx.x * 128 + (wid & 3) * 32;
  int ntn = N >> 4;
  int nt0 = n0 >> 4;

  v8f acc[2][2] = {{ {0.f}, {0.f} }, { {0.f}, {0.f} }};

  for (int k0 = 0; k0 < K; k0 += 32) {
    int kt = k0 >> 5;
    v16bf a0 = frag_a(A, K, m0,      k0, lane);
    v16bf a1 = frag_a(A, K, m0 + 16, k0, lane);
    if (k0 + 32 < K)
      __builtin_prefetch(A + (long)(m0 + (lane & 15)) * K + k0 + 32, 0, 0);
    v16bf b0 = frag_b_packed(Bp, ntn, kt, nt0,     lane);
    v16bf b1 = frag_b_packed(Bp, ntn, kt, nt0 + 1, lane);
    acc[0][0] = wmma_bf16(a0, b0, acc[0][0]);
    acc[0][1] = wmma_bf16(a0, b1, acc[0][1]);
    acc[1][0] = wmma_bf16(a1, b0, acc[1][0]);
    acc[1][1] = wmma_bf16(a1, b1, acc[1][1]);
  }

  int c = lane & 15;
  int msel = (lane >> 4) * 8;
#pragma unroll
  for (int mi = 0; mi < 2; ++mi)
#pragma unroll
    for (int ni = 0; ni < 2; ++ni)
#pragma unroll
      for (int r = 0; r < 8; ++r) {
        int row = m0 + mi * 16 + msel + r;
        int col = n0 + ni * 16 + c;
        D[(long)row * N + col] = acc[mi][ni][r];
      }
}

// ---------------- combine ----------------

__global__ void __launch_bounds__(256)
combine_kernel(const float* __restrict__ x, const float* __restrict__ down,
               const float* __restrict__ fdown,
               const int* __restrict__ e_flat, const int* __restrict__ pos,
               const float* __restrict__ w_flat, const int* __restrict__ ovf,
               float* __restrict__ out) {
  int t = blockIdx.x;
  int e0 = e_flat[2 * t],     e1 = e_flat[2 * t + 1];
  int p0 = pos[2 * t],        p1 = pos[2 * t + 1];
  float w0 = w_flat[2 * t],   w1 = w_flat[2 * t + 1];
  int is_ovf = ovf[t];
  const float* xr = x + (long)t * H_DIM;
  float* orow = out + (long)t * H_DIM;
  const float* d0 = (p0 >= 0) ? down + ((long)e0 * CAP + p0) * H_DIM : nullptr;
  const float* d1 = (p1 >= 0) ? down + ((long)e1 * CAP + p1) * H_DIM : nullptr;
  const float* fd = fdown + (long)t * H_DIM;
  for (int h = threadIdx.x; h < H_DIM; h += 256) {
    float xv = xr[h];
    float o;
    if (is_ovf) {
      o = fd[h] + xv;                          // fallback FFN output + residual
    } else {
      o = xv;
      o += w0 * (d0[h] + xv);                  // expert_out = down + buf(=x)
      o += w1 * (d1[h] + xv);
    }
    orow[h] = o;
  }
}

// ---------------- host launch ----------------

extern "C" void kernel_launch(void* const* d_in, const int* in_sizes, int n_in,
                              void* d_out, int out_size, void* d_ws, size_t ws_size,
                              hipStream_t stream) {
  const float* x      = (const float*)d_in[0];
  const float* Wr     = (const float*)d_in[1];
  const float* norm_w = (const float*)d_in[2];
  const float* Wg     = (const float*)d_in[3];
  const float* Wu     = (const float*)d_in[4];
  const float* Wd     = (const float*)d_in[5];
  const float* f_norm = (const float*)d_in[6];
  const float* fWg    = (const float*)d_in[7];
  const float* fWu    = (const float*)d_in[8];
  const float* fWd    = (const float*)d_in[9];
  float* out = (float*)d_out;

  char* wsp = (char*)d_ws;
  auto alloc = [&](size_t bytes) -> char* {
    char* p = wsp;
    wsp += (bytes + 255) & ~(size_t)255;
    return p;
  };

  __bf16* Wg_p   = (__bf16*)alloc((size_t)NEXP * H_DIM * F_DIM * 2);
  __bf16* Wu_p   = (__bf16*)alloc((size_t)NEXP * H_DIM * F_DIM * 2);
  __bf16* Wd_p   = (__bf16*)alloc((size_t)NEXP * F_DIM * H_DIM * 2);
  __bf16* fWg_p  = (__bf16*)alloc((size_t)H_DIM * F4_DIM * 2);
  __bf16* fWu_p  = (__bf16*)alloc((size_t)H_DIM * F4_DIM * 2);
  __bf16* fWd_p  = (__bf16*)alloc((size_t)F4_DIM * H_DIM * 2);
  int*    e_flat = (int*)alloc((size_t)NASSIGN * 4);
  float*  w_flat = (float*)alloc((size_t)NASSIGN * 4);
  int*    pos    = (int*)alloc((size_t)NASSIGN * 4);
  int*    blkhist= (int*)alloc((size_t)NBLK * NEXP * 4);
  int*    blkbase= (int*)alloc((size_t)NBLK * NEXP * 4);
  int*    ovf    = (int*)alloc((size_t)TOKENS * 4);
  __bf16* hbuf   = (__bf16*)alloc((size_t)NEXP * CAP * H_DIM * 2);
  __bf16* act    = (__bf16*)alloc((size_t)NEXP * CAP * F_DIM * 2);
  float*  down   = (float*)alloc((size_t)NEXP * CAP * H_DIM * 4);
  __bf16* fh     = (__bf16*)alloc((size_t)TOKENS * H_DIM * 2);
  __bf16* fact   = (__bf16*)alloc((size_t)TOKENS * F4_DIM * 2);
  float*  fdown  = (float*)alloc((size_t)TOKENS * H_DIM * 4);

  // 1) weight pre-swizzle into WMMA B-fragment layout (fuses f32->bf16)
  {
    long dwe = (long)H_DIM * F_DIM / 2;                   // dwords per expert
    pack_b_kernel<<<dim3((dwe + 255) / 256, 1, NEXP), 256, 0, stream>>>(
        Wg, Wg_p, F_DIM, H_DIM, (long)H_DIM * F_DIM, (long)H_DIM * F_DIM);
    pack_b_kernel<<<dim3((dwe + 255) / 256, 1, NEXP), 256, 0, stream>>>(
        Wu, Wu_p, F_DIM, H_DIM, (long)H_DIM * F_DIM, (long)H_DIM * F_DIM);
    long dwd = (long)F_DIM * H_DIM / 2;
    pack_b_kernel<<<dim3((dwd + 255) / 256, 1, NEXP), 256, 0, stream>>>(
        Wd, Wd_p, H_DIM, F_DIM, (long)F_DIM * H_DIM, (long)F_DIM * H_DIM);
    long dwf = (long)H_DIM * F4_DIM / 2;
    pack_b_kernel<<<dim3((dwf + 255) / 256, 1, 1), 256, 0, stream>>>(
        fWg, fWg_p, F4_DIM, H_DIM, 0, 0);
    pack_b_kernel<<<dim3((dwf + 255) / 256, 1, 1), 256, 0, stream>>>(
        fWu, fWu_p, F4_DIM, H_DIM, 0, 0);
    pack_b_kernel<<<dim3((dwf + 255) / 256, 1, 1), 256, 0, stream>>>(
        fWd, fWd_p, H_DIM, F4_DIM, 0, 0);
  }

  // 2) router
  router_kernel<<<dim3(TOKENS / 256), 256, 0, stream>>>(x, Wr, e_flat, w_flat);

  // 3) deterministic capacity assignment (order-preserving scan)
  hist_kernel<<<dim3(NBLK), 256, 0, stream>>>(e_flat, blkhist);
  scan_kernel<<<dim3(1), 32, 0, stream>>>(blkhist, blkbase);
  assign_kernel<<<dim3(NBLK), 256, 0, stream>>>(e_flat, blkbase, pos);
  overflow_kernel<<<dim3(TOKENS / 256), 256, 0, stream>>>(pos, ovf);

  // 4) RMSNorm + scatter into expert buffers; fallback RMSNorm
  rms_scatter_kernel<<<dim3(NASSIGN), 256, 0, stream>>>(x, norm_w, e_flat, pos, hbuf);
  rms_fallback_kernel<<<dim3(TOKENS), 256, 0, stream>>>(x, f_norm, fh);

  // 5) expert gate/up GEMM + GELU*up epilogue:  [C x H] @ [H x F] per expert
  gemm_glu_kernel<<<dim3(F_DIM / 128, CAP / 64, NEXP), 256, 0, stream>>>(
      hbuf, Wg_p, Wu_p, act, F_DIM, H_DIM,
      (long)CAP * H_DIM, (long)H_DIM * F_DIM, (long)CAP * F_DIM);

  // 6) expert down GEMM: [C x F] @ [F x H] per expert
  gemm_f32out_kernel<<<dim3(H_DIM / 128, CAP / 64, NEXP), 256, 0, stream>>>(
      act, Wd_p, down, H_DIM, F_DIM,
      (long)CAP * F_DIM, (long)F_DIM * H_DIM, (long)CAP * H_DIM);

  // 7) fallback gate/up GEMM: [N x H] @ [H x F4]
  gemm_glu_kernel<<<dim3(F4_DIM / 128, TOKENS / 64, 1), 256, 0, stream>>>(
      fh, fWg_p, fWu_p, fact, F4_DIM, H_DIM, 0, 0, 0);

  // 8) fallback down GEMM: [N x F4] @ [F4 x H]
  gemm_f32out_kernel<<<dim3(H_DIM / 128, TOKENS / 64, 1), 256, 0, stream>>>(
      fact, fWd_p, fdown, H_DIM, F4_DIM, 0, 0, 0);

  // 9) combine
  combine_kernel<<<dim3(TOKENS), 256, 0, stream>>>(
      x, down, fdown, e_flat, pos, w_flat, ovf, out);

  (void)in_sizes; (void)n_in; (void)out_size; (void)ws_size;
}
